// EuclideanCodebook_7378753815010
// MI455X (gfx1250) — matrix operand order repaired and checked by reading, not compiled, not run on hardware
//
#include <hip/hip_runtime.h>
#include <hip/hip_bf16.h>

#define DECAY 0.1f
#define EPS_ 1e-5f

#define N_ 262144
#define K_ 1024
#define D_ 64

typedef __attribute__((ext_vector_type(16))) __bf16          v16bf;
typedef __attribute__((ext_vector_type(8)))  float           v8f;
typedef __attribute__((ext_vector_type(8)))  unsigned short  us8;
typedef __attribute__((ext_vector_type(4)))  int             v4i;

#if defined(__has_builtin)
#if __has_builtin(__builtin_amdgcn_global_load_async_to_lds_b128)
#define HAVE_ASYNC_LDS 1
#endif
#if __has_builtin(__builtin_amdgcn_s_wait_asynccnt)
#define HAVE_WAIT_ASYNC 1
#endif
#endif

// Round-to-nearest-even fp32 -> bf16 (bit-level)
static __device__ __forceinline__ unsigned short f2bf_bits(float f) {
    union { float f; unsigned int u; } v; v.f = f;
    unsigned int r = v.u + 0x7FFFu + ((v.u >> 16) & 1u);
    return (unsigned short)(r >> 16);
}

union V16U { v16bf v; unsigned short s[16]; us8 p[2]; };

// ---------------------------------------------------------------------------
// Kernel 0: zero accumulators, pre-pack embed->bf16, precompute e2[K]
//   65536 threads: i in [0, K*D)
// ---------------------------------------------------------------------------
__global__ __launch_bounds__(256) void vq_prep_kernel(
    const float* __restrict__ embed, float* __restrict__ accum /* esum+bins */,
    unsigned short* __restrict__ ebf, float* __restrict__ e2w)
{
    const int i = blockIdx.x * 256 + threadIdx.x;      // [0, 65536)
    accum[i] = 0.0f;                                   // esum part
    if (i < K_) accum[K_ * D_ + i] = 0.0f;             // bins part
    ebf[i] = f2bf_bits(embed[i]);
    if (i < K_) {
        const float4* r4 = (const float4*)(embed + (size_t)i * D_);
        float s = 0.0f;
#pragma unroll
        for (int j = 0; j < D_ / 4; ++j) {
            float4 v = r4[j];
            s += v.x * v.x + v.y * v.y + v.z * v.z + v.w * v.w;
        }
        e2w[i] = s;
    }
}

// ---------------------------------------------------------------------------
// Kernel 1: distance GEMM (bf16 WMMA) + argmin + gather + scatter-add
//   block = 256 threads = 8 waves; each wave owns TWO 16-row x tiles (32 rows)
//   -> two independent WMMA accumulation chains per wave
// ---------------------------------------------------------------------------
__global__ __launch_bounds__(256) void vq_main_kernel(
    const float* __restrict__ x, const float* __restrict__ embed,
    const unsigned short* __restrict__ ebf, const float* __restrict__ e2w,
    float* __restrict__ outQuant, float* __restrict__ outInd,
    float* __restrict__ bins, float* __restrict__ embedSum)
{
    __shared__ __attribute__((aligned(16))) unsigned short embLds[128 * D_]; // 16 KB
    __shared__ __attribute__((aligned(16))) float e2Lds[128];
    __shared__ int idxLds[8][32];

    const int tid  = threadIdx.x;
    const int wave = tid >> 5;
    const int l    = tid & 31;
    const int half = l >> 4;
    const int m    = l & 15;
    const int rowBase = blockIdx.x * 256 + wave * 32;

    // ---- Build A matrices for both row tiles (16x32 bf16 each, D=64 => 2 K-tiles)
    // ISA layout per lane(half,m): elems 0..7 <- K = half*8+0..7 ; 8..15 <- K = 16+half*8+0..7
    V16U a0A, a1A, a0B, a1B;
    {
        const float* xrA = x + (size_t)(rowBase + m) * D_;
        const float* xrB = xrA + 16 * D_;
        const int c0 = half * 8;
        const int c1 = 16 + half * 8;
#pragma unroll
        for (int e = 0; e < 8; ++e) {
            a0A.s[e]     = f2bf_bits(xrA[c0 + e]);
            a0A.s[8 + e] = f2bf_bits(xrA[c1 + e]);
            a1A.s[e]     = f2bf_bits(xrA[32 + c0 + e]);
            a1A.s[8 + e] = f2bf_bits(xrA[32 + c1 + e]);
            a0B.s[e]     = f2bf_bits(xrB[c0 + e]);
            a0B.s[8 + e] = f2bf_bits(xrB[c1 + e]);
            a1B.s[e]     = f2bf_bits(xrB[32 + c0 + e]);
            a1B.s[8 + e] = f2bf_bits(xrB[32 + c1 + e]);
        }
    }

    float bV[16], bI[16];   // [0..7] tile A rows, [8..15] tile B rows
#pragma unroll
    for (int r = 0; r < 16; ++r) { bV[r] = 3.4e38f; bI[r] = 0.0f; }

    for (int chunk = 0; chunk < K_ / 128; ++chunk) {
        __syncthreads();
        // ---- Stage 16 KB bf16 codebook chunk + 512 B e2 chunk into LDS ----
        {
            const us8* gsrc = (const us8*)(ebf + (size_t)chunk * 128 * D_);
#ifdef HAVE_ASYNC_LDS
#pragma unroll
            for (int rr = 0; rr < 4; ++rr) {
                const int off = tid + rr * 256;     // 16-byte units
                __builtin_amdgcn_global_load_async_to_lds_b128(
                    (v4i*)(gsrc + off),
                    (v4i*)((char*)embLds + (size_t)off * 16), 0, 0);
            }
            if (tid < 32) {
                __builtin_amdgcn_global_load_async_to_lds_b128(
                    (v4i*)((const float4*)(e2w + chunk * 128) + tid),
                    (v4i*)((char*)e2Lds + (size_t)tid * 16), 0, 0);
            }
#ifdef HAVE_WAIT_ASYNC
            __builtin_amdgcn_s_wait_asynccnt(0);
#else
            asm volatile("s_wait_asynccnt 0" ::: "memory");
#endif
#else
#pragma unroll
            for (int rr = 0; rr < 4; ++rr) {
                const int off = tid + rr * 256;
                ((us8*)embLds)[off] = gsrc[off];
            }
            if (tid < 128) e2Lds[tid] = e2w[chunk * 128 + tid];
#endif
        }
        __syncthreads();

        for (int tile = 0; tile < 8; ++tile) {
            const int nloc = tile * 16 + m; // code within chunk for this lane's column
            // B layout per lane(half,m): elems 0..7 <- K = half*8+0..7 ; 8..15 <- K = 16+half*8+0..7
            const unsigned short* brow = &embLds[nloc * D_ + half * 8];
            V16U b0, b1;
            b0.p[0] = *(const us8*)(brow);
            b0.p[1] = *(const us8*)(brow + 16);
            b1.p[0] = *(const us8*)(brow + 32);
            b1.p[1] = *(const us8*)(brow + 48);

            v8f accA = {}, accB = {};
            accA = __builtin_amdgcn_wmma_f32_16x16x32_bf16(false, a0A.v, false, b0.v,
                                                           (short)0, accA, false, false);
            accB = __builtin_amdgcn_wmma_f32_16x16x32_bf16(false, a0B.v, false, b0.v,
                                                           (short)0, accB, false, false);
            accA = __builtin_amdgcn_wmma_f32_16x16x32_bf16(false, a1A.v, false, b1.v,
                                                           (short)0, accA, false, false);
            accB = __builtin_amdgcn_wmma_f32_16x16x32_bf16(false, a1B.v, false, b1.v,
                                                           (short)0, accB, false, false);

            const float e2v  = e2Lds[nloc];
            const float idxf = (float)(chunk * 128 + nloc);
#pragma unroll
            for (int r = 0; r < 8; ++r) {
                float dA = fmaf(-2.0f, accA[r], e2v);
                float dB = fmaf(-2.0f, accB[r], e2v);
                if (dA < bV[r])     { bV[r] = dA;     bI[r] = idxf; }
                if (dB < bV[8 + r]) { bV[8 + r] = dB; bI[8 + r] = idxf; }
            }
        }
    }

    // ---- Reduce argmin across the 16 lanes sharing each row set ----
#pragma unroll
    for (int off = 1; off < 16; off <<= 1) {
#pragma unroll
        for (int r = 0; r < 16; ++r) {
            float ov = __shfl_xor(bV[r], off, 32);
            float oi = __shfl_xor(bI[r], off, 32);
            if (ov < bV[r] || (ov == bV[r] && oi < bI[r])) { bV[r] = ov; bI[r] = oi; }
        }
    }
    // lanes 0..15 hold rows M=0..7; lanes 16..31 hold rows M=8..15 (per tile)
    if (m == 0) {
#pragma unroll
        for (int r = 0; r < 8; ++r) {
            const int rowA = half * 8 + r;          // tile A
            const int rowB = 16 + half * 8 + r;     // tile B
            const int idxA = (int)bI[r];
            const int idxB = (int)bI[8 + r];
            idxLds[wave][rowA] = idxA;
            idxLds[wave][rowB] = idxB;
            outInd[rowBase + rowA] = (float)idxA;
            outInd[rowBase + rowB] = (float)idxB;
        }
    }
    __syncthreads();

    // ---- Gather quantize + scatter-add embed_sum / bins ----
    for (int r = 0; r < 32; ++r) {
        const int idx  = idxLds[wave][r];
        const int grow = rowBase + r;
        const float2* ep = (const float2*)(embed + (size_t)idx * D_);
        const float2* xp = (const float2*)(x + (size_t)grow * D_);
        float2 q  = ep[l];
        float2 xv = xp[l];
        ((float2*)(outQuant + (size_t)grow * D_))[l] = q;
        atomicAdd(&embedSum[idx * D_ + 2 * l + 0], xv.x);
        atomicAdd(&embedSum[idx * D_ + 2 * l + 1], xv.y);
        if (l == 0) atomicAdd(&bins[idx], 1.0f);
    }
}

// ---------------------------------------------------------------------------
// Kernel 2: EMA update + Laplace smoothing + normalization (one block, K=1024)
// ---------------------------------------------------------------------------
__global__ __launch_bounds__(1024) void vq_finalize_kernel(
    const float* __restrict__ clusterSize, const float* __restrict__ embedAvg,
    const float* __restrict__ bins, const float* __restrict__ embedSum,
    float* __restrict__ outCs, float* __restrict__ outAvg, float* __restrict__ outNorm)
{
    __shared__ float red[1024];
    __shared__ float csS[1024];
    const int k = threadIdx.x;

    float csn = clusterSize[k] * DECAY + bins[k] * (1.0f - DECAY);
    outCs[k] = csn;
    red[k]   = csn;
    __syncthreads();
#pragma unroll
    for (int off = 512; off > 0; off >>= 1) {
        if (k < off) red[k] += red[k + off];
        __syncthreads();
    }
    const float n = red[0];
    csS[k] = (csn + EPS_) / (n + (float)K_ * EPS_) * n;
    __syncthreads();

    for (int i = k; i < K_ * D_; i += 1024) {
        float a = embedAvg[i] * DECAY + embedSum[i] * (1.0f - DECAY);
        outAvg[i]  = a;
        outNorm[i] = a / csS[i >> 6];
    }
}

// ---------------------------------------------------------------------------
extern "C" void kernel_launch(void* const* d_in, const int* in_sizes, int n_in,
                              void* d_out, int out_size, void* d_ws, size_t ws_size,
                              hipStream_t stream) {
    (void)in_sizes; (void)n_in; (void)out_size; (void)ws_size;
    const float* x      = (const float*)d_in[0];
    const float* embed  = (const float*)d_in[1];
    const float* cs     = (const float*)d_in[2];
    const float* eavg   = (const float*)d_in[3];

    float* out      = (float*)d_out;
    float* outQuant = out;                                   // [N, D]
    float* outInd   = outQuant + (size_t)N_ * D_;            // [N]
    float* outCs    = outInd + N_;                           // [K]
    float* outAvg   = outCs + K_;                            // [K, D]
    float* outNorm  = outAvg + (size_t)K_ * D_;              // [K, D]

    // workspace layout: esum[K*D] f32 | bins[K] f32 | ebf[K*D] u16 | e2[K] f32
    float* accum = (float*)d_ws;
    float* esum  = accum;                      // [K*D]
    float* binsp = accum + (size_t)K_ * D_;    // [K]
    unsigned short* ebf = (unsigned short*)(binsp + K_);     // [K*D] bf16 bits
    float* e2w = (float*)(ebf + (size_t)K_ * D_);            // [K]

    vq_prep_kernel<<<K_ * D_ / 256, 256, 0, stream>>>(embed, accum, ebf, e2w);
    vq_main_kernel<<<N_ / 256, 256, 0, stream>>>(x, embed, ebf, e2w,
                                                 outQuant, outInd, binsp, esum);
    vq_finalize_kernel<<<1, 1024, 0, stream>>>(cs, eavg, binsp, esum,
                                               outCs, outAvg, outNorm);
}
